// ViT_47184510714573
// MI455X (gfx1250) — 16-bit and 8-bit matrix operand order repaired and checked by reading, compiled, not run on hardware
//
#include <hip/hip_runtime.h>
#include <hip/hip_fp16.h>
#include <cstdint>
#include <math.h>

typedef __attribute__((ext_vector_type(8)))  int      v8i;
typedef __attribute__((ext_vector_type(8)))  float    v8f;
typedef __attribute__((ext_vector_type(16))) _Float16 v16h;
typedef __attribute__((ext_vector_type(4)))  unsigned v4u;
typedef __attribute__((ext_vector_type(4)))  int      v4i;

#define BATCH   32
#define SEQ     197
#define EMB     768
#define NHEAD   12
#define HDIM    64
#define HID     3072
#define NCLS    1000
#define NPATCH  196
#define MTOT    (BATCH*SEQ)
#define MPATCH  (BATCH*NPATCH)
#define SQ      208
#define SK      224
#define BH      (BATCH*NHEAD)

#if __has_builtin(__builtin_amdgcn_tensor_load_to_lds) && __has_builtin(__builtin_amdgcn_s_wait_tensorcnt)
#define USE_TDM 1
#else
#define USE_TDM 0
#endif

__global__ void k_zero1(unsigned* p) { p[0] = 0u; }

__global__ void k_absmax(const float* __restrict__ x, long long n, unsigned* __restrict__ out) {
    __shared__ float red[256];
    float m = 0.f;
    for (long long i = (long long)blockIdx.x * 256 + threadIdx.x; i < n;
         i += (long long)gridDim.x * 256)
        m = fmaxf(m, fabsf(x[i]));
    red[threadIdx.x] = m;
    __syncthreads();
    for (int s = 128; s > 0; s >>= 1) {
        if ((int)threadIdx.x < s) red[threadIdx.x] = fmaxf(red[threadIdx.x], red[threadIdx.x + s]);
        __syncthreads();
    }
    if (threadIdx.x == 0) atomicMax(out, __float_as_uint(red[0]));
}

__global__ void k_quant(const float* __restrict__ x, long long n,
                        const unsigned* __restrict__ amax, float* __restrict__ scaleOut,
                        int8_t* __restrict__ q) {
    float am = __uint_as_float(amax[0]);
    float s  = fmaxf(am / 127.f, 1e-8f);
    if (blockIdx.x == 0 && threadIdx.x == 0) scaleOut[0] = s;
    float inv = 1.f / s;
    for (long long i = (long long)blockIdx.x * 256 + threadIdx.x; i < n;
         i += (long long)gridDim.x * 256) {
        float v = rintf(x[i] * inv);
        v = fminf(fmaxf(v, -127.f), 127.f);
        q[i] = (int8_t)v;
    }
}

__global__ void k_quantT(const float* __restrict__ w, int K, int N,
                         const unsigned* __restrict__ amax, float* __restrict__ scaleOut,
                         int8_t* __restrict__ qT) {
    float am = __uint_as_float(amax[0]);
    float s  = fmaxf(am / 127.f, 1e-8f);
    if (blockIdx.x == 0 && threadIdx.x == 0) scaleOut[0] = s;
    float inv = 1.f / s;
    long long total = (long long)K * N;
    for (long long i = (long long)blockIdx.x * 256 + threadIdx.x; i < total;
         i += (long long)gridDim.x * 256) {
        int  n = (int)(i % N);
        long long k = i / N;
        float v = rintf(w[i] * inv);
        v = fminf(fmaxf(v, -127.f), 127.f);
        qT[(size_t)n * K + k] = (int8_t)v;
    }
}

__global__ void k_f32tof16(const float* __restrict__ x, _Float16* __restrict__ y, long long n) {
    for (long long i = (long long)blockIdx.x * 256 + threadIdx.x; i < n;
         i += (long long)gridDim.x * 256)
        y[i] = (_Float16)x[i];
}

#if USE_TDM
__device__ __forceinline__ void tdm_load_2d(unsigned ldsAddr, const void* gptr,
                                            unsigned dim0_rem, unsigned dim1_rem,
                                            unsigned stride, unsigned tile0, unsigned tile1)
{
    unsigned long long ga = (unsigned long long)gptr;
    v4u g0;
    g0[0] = 1u;
    g0[1] = ldsAddr;
    g0[2] = (unsigned)ga;
    g0[3] = ((unsigned)(ga >> 32) & 0x01FFFFFFu) | 0x80000000u;
    v8i g1;
    g1[0] = 0;
    g1[1] = (int)((dim0_rem & 0xFFFFu) << 16);
    g1[2] = (int)((dim0_rem >> 16) | ((dim1_rem & 0xFFFFu) << 16));
    g1[3] = (int)((dim1_rem >> 16) | (tile0 << 16));
    g1[4] = (int)tile1;
    g1[5] = (int)stride;
    g1[6] = 0;
    g1[7] = 0;
    v4i z4 = {0, 0, 0, 0};
    v8i z8 = {0, 0, 0, 0, 0, 0, 0, 0};
    __builtin_amdgcn_tensor_load_to_lds(g0, g1, z4, z4, z8, 0);
}
#endif

__global__ __launch_bounds__(256) void k_gemm_i8(
    const int8_t* __restrict__ A,
    const int8_t* __restrict__ BT,
    const float* __restrict__ sA, const float* __restrict__ sB,
    const float* __restrict__ bias,
    const float* resid,
    float* C,
    int M, int N, int K, int op)
{
    __shared__ __align__(16) int8_t sa [2][128 * 64];
    __shared__ __align__(16) int8_t sbt[2][128 * 64];
    const int tid = threadIdx.x;
    const int m0  = blockIdx.y * 128;
    const int n0  = blockIdx.x * 128;
    const int w   = tid >> 5, lane = tid & 31;
    const int wm  = w >> 1,   wn   = w & 1;
    const int lhi = lane >> 4, llo = lane & 15;
    const int nt  = K >> 6;

    v8i acc[2][4] = {};

#if USE_TDM
    auto issue = [&](int t, int buf) {
        int k0 = t << 6;
        tdm_load_2d((unsigned)(size_t)&sa[buf][0],  A  + (size_t)m0 * K + k0,
                    (unsigned)(K - k0), (unsigned)(M - m0), (unsigned)K, 64u, 128u);
        tdm_load_2d((unsigned)(size_t)&sbt[buf][0], BT + (size_t)n0 * K + k0,
                    (unsigned)(K - k0), (unsigned)(N - n0), (unsigned)K, 64u, 128u);
    };
    if (tid < 32) issue(0, 0);
#endif

    for (int t = 0; t < nt; ++t) {
        const int buf = t & 1;
#if USE_TDM
        if (t + 1 < nt) {
            if (tid < 32) issue(t + 1, buf ^ 1);
            __builtin_amdgcn_s_wait_tensorcnt(2);
        } else {
            __builtin_amdgcn_s_wait_tensorcnt(0);
        }
        __syncthreads();
#else
        const int k0 = t << 6;
        __syncthreads();
        #pragma unroll
        for (int it = 0; it < 2; ++it) {
            int idx = tid + it * 256;
            int row = idx >> 2;
            int co  = (idx & 3) << 4;
            int gr  = m0 + row;
            int4 val = make_int4(0, 0, 0, 0);
            if (gr < M) val = *(const int4*)(A + (size_t)gr * K + k0 + co);
            *(int4*)(&sa[buf][row * 64 + co]) = val;
        }
        #pragma unroll
        for (int it = 0; it < 2; ++it) {
            int idx = tid + it * 256;
            int row = idx >> 2;
            int co  = (idx & 3) << 4;
            *(int4*)(&sbt[buf][row * 64 + co]) =
                *(const int4*)(BT + (size_t)(n0 + row) * K + k0 + co);
        }
        __syncthreads();
#endif

        v8i af[2], bf[4];
        #pragma unroll
        for (int mi = 0; mi < 2; ++mi) {
            const unsigned* p = (const unsigned*)(&sa[buf][(wm * 32 + mi * 16 + llo) * 64]);
            #pragma unroll
            for (int i = 0; i < 8; ++i) {
                int dw = ((i >> 2) << 3) + (lhi << 2) + (i & 3);
                af[mi][i] = (int)p[dw];
            }
        }
        #pragma unroll
        for (int ni = 0; ni < 4; ++ni) {
            const unsigned* p = (const unsigned*)(&sbt[buf][(wn * 64 + ni * 16 + llo) * 64]);
            #pragma unroll
            for (int i = 0; i < 8; ++i) {
                int dw = ((i >> 2) << 3) + (lhi << 2) + (i & 3);
                bf[ni][i] = (int)p[dw];
            }
        }
        #pragma unroll
        for (int mi = 0; mi < 2; ++mi)
            #pragma unroll
            for (int ni = 0; ni < 4; ++ni)
                acc[mi][ni] = __builtin_amdgcn_wmma_i32_16x16x64_iu8(
                    true, af[mi], true, bf[ni], acc[mi][ni], false, false);

        __syncthreads();
    }

    const float s = sA[0] * sB[0];
    #pragma unroll
    for (int mi = 0; mi < 2; ++mi) {
        #pragma unroll
        for (int ni = 0; ni < 4; ++ni) {
            int col = n0 + wn * 64 + ni * 16 + llo;
            float bv = bias[col];
            #pragma unroll
            for (int r = 0; r < 8; ++r) {
                int row = m0 + wm * 32 + mi * 16 + r + lhi * 8;
                if (row < M) {
                    float v = (float)acc[mi][ni][r] * s + bv;
                    if (op == 1) v = 0.5f * v * (1.f + erff(v * 0.70710678118654752f));
                    else if (op == 2) v += resid[(size_t)row * N + col];
                    C[(size_t)row * N + col] = v;
                }
            }
        }
    }
}

__global__ __launch_bounds__(32) void k_gemm_f16(
    const _Float16* __restrict__ A, const _Float16* __restrict__ BT,
    const float* __restrict__ bias, float* __restrict__ C,
    int M, int N, int K, int lda, int ldb, int ldc,
    long long aStride, long long bStride,
    long long cOuterStride, long long cInnerStride, int cInner)
{
    const int z = blockIdx.z;
    const _Float16* Ab = A  + (size_t)z * aStride;
    const _Float16* Bb = BT + (size_t)z * bStride;
    float* Cb = C + (size_t)(z / cInner) * cOuterStride + (size_t)(z % cInner) * cInnerStride;
    const int m0 = blockIdx.y * 16, n0 = blockIdx.x * 16;
    const int lane = threadIdx.x, lhi = lane >> 4, llo = lane & 15;

    v8f acc = {};
    const unsigned* pa = (const unsigned*)(Ab + (size_t)(m0 + llo) * lda);
    const unsigned* pb = (const unsigned*)(Bb + (size_t)(n0 + llo) * ldb);
    for (int k0 = 0; k0 < K; k0 += 32) {
        union { v16h v; unsigned u[8]; } a, b;
        const unsigned* qa = pa + (k0 >> 1);
        const unsigned* qb = pb + (k0 >> 1);
        #pragma unroll
        for (int i = 0; i < 8; ++i) {
            a.u[i] = qa[i + (lhi << 3)];
            b.u[i] = qb[i + (lhi << 3)];
        }
        if (k0 + 32 < K) {
            __builtin_prefetch(qa + (k0 >> 1) + 16, 0, 1);
            __builtin_prefetch(qb + (k0 >> 1) + 16, 0, 1);
        }
        acc = __builtin_amdgcn_wmma_f32_16x16x32_f16(
            false, a.v, false, b.v, (short)0, acc, false, false);
    }

    const int col = n0 + llo;
    #pragma unroll
    for (int r = 0; r < 8; ++r) {
        int row = m0 + r + lhi * 8;
        if (row < M && col < N) {
            float v = acc[r];
            if (bias) v += bias[col];
            Cb[(size_t)row * ldc + col] = v;
        }
    }
}

__global__ __launch_bounds__(256) void k_layernorm(
    const float* __restrict__ x, const float* __restrict__ g, const float* __restrict__ b,
    float* __restrict__ y, long long inStride, long long outStride)
{
    __shared__ float s1[256], s2[256];
    const float* xr = x + (size_t)blockIdx.x * inStride;
    float* yr = y + (size_t)blockIdx.x * outStride;
    float a = 0.f, q = 0.f;
    for (int i = threadIdx.x; i < EMB; i += 256) { float v = xr[i]; a += v; q += v * v; }
    s1[threadIdx.x] = a; s2[threadIdx.x] = q;
    __syncthreads();
    for (int s = 128; s > 0; s >>= 1) {
        if ((int)threadIdx.x < s) { s1[threadIdx.x] += s1[threadIdx.x + s]; s2[threadIdx.x] += s2[threadIdx.x + s]; }
        __syncthreads();
    }
    float mu  = s1[0] * (1.f / EMB);
    float var = s2[0] * (1.f / EMB) - mu * mu;
    float rst = rsqrtf(var + 1e-5f);
    for (int i = threadIdx.x; i < EMB; i += 256)
        yr[i] = (xr[i] - mu) * rst * g[i] + b[i];
}

__global__ __launch_bounds__(64) void k_softmax(const float* __restrict__ S, _Float16* __restrict__ P) {
    __shared__ float red[64];
    const int row = blockIdx.x;
    const float* sr = S + (size_t)row * SK;
    _Float16* pr = P + (size_t)row * SK;
    const int t = threadIdx.x;
    const int r = row % SQ;
    if (r >= SEQ) {
        for (int c = t; c < SK; c += 64) pr[c] = (_Float16)0.f;
        return;
    }
    float mx = -1e30f;
    for (int c = t; c < SEQ; c += 64) mx = fmaxf(mx, sr[c]);
    red[t] = mx; __syncthreads();
    for (int s = 32; s > 0; s >>= 1) { if (t < s) red[t] = fmaxf(red[t], red[t + s]); __syncthreads(); }
    mx = red[0]; __syncthreads();
    float sum = 0.f;
    for (int c = t; c < SEQ; c += 64) sum += __expf(sr[c] - mx);
    red[t] = sum; __syncthreads();
    for (int s = 32; s > 0; s >>= 1) { if (t < s) red[t] += red[t + s]; __syncthreads(); }
    float inv = 1.f / red[0];
    for (int c = t; c < SEQ; c += 64) pr[c] = (_Float16)(__expf(sr[c] - mx) * inv);
    for (int c = SEQ + t; c < SK; c += 64) pr[c] = (_Float16)0.f;
}

__global__ void k_patchify(const float* __restrict__ x, _Float16* __restrict__ p) {
    long long total = (long long)MPATCH * EMB;
    for (long long i = (long long)blockIdx.x * 256 + threadIdx.x; i < total;
         i += (long long)gridDim.x * 256) {
        int e = (int)(i % EMB);
        long long rp = i / EMB;
        int b  = (int)(rp / NPATCH);
        int pt = (int)(rp % NPATCH);
        int gy = pt / 14, gx = pt % 14;
        int c  = e >> 8;
        int rm = e & 255;
        int py = rm >> 4, px = rm & 15;
        p[i] = (_Float16)x[(((long long)b * 3 + c) * 224 + gy * 16 + py) * 224 + gx * 16 + px];
    }
}

__global__ void k_assemble(const float* __restrict__ pe, const float* __restrict__ cls,
                           const float* __restrict__ pos, float* __restrict__ h) {
    long long total = (long long)MTOT * EMB;
    for (long long i = (long long)blockIdx.x * 256 + threadIdx.x; i < total;
         i += (long long)gridDim.x * 256) {
        int e = (int)(i % EMB);
        long long r = i / EMB;
        int b = (int)(r / SEQ);
        int s = (int)(r % SEQ);
        float v = (s == 0) ? cls[e] : pe[((long long)b * NPATCH + (s - 1)) * EMB + e];
        h[i] = v + pos[(long long)s * EMB + e];
    }
}

__global__ void k_qkv_split(const float* __restrict__ qkv,
                            _Float16* __restrict__ Q, _Float16* __restrict__ Kh,
                            _Float16* __restrict__ Vt) {
    long long total = (long long)MTOT * EMB;
    for (long long i = (long long)blockIdx.x * 256 + threadIdx.x; i < total;
         i += (long long)gridDim.x * 256) {
        int e = (int)(i % EMB);
        long long r = i / EMB;
        int b = (int)(r / SEQ);
        int s = (int)(r % SEQ);
        int h = e >> 6, d = e & 63;
        const float* src = qkv + r * (3 * EMB);
        int bh = b * NHEAD + h;
        Q [((size_t)bh * SQ + s) * HDIM + d] = (_Float16)(src[e] * 0.125f);
        Kh[((size_t)bh * SK + s) * HDIM + d] = (_Float16)src[EMB + e];
        Vt[((size_t)bh * HDIM + d) * SK + s] = (_Float16)src[2 * EMB + e];
    }
}

__global__ void k_head(const float* __restrict__ cls, const float* __restrict__ w,
                       const float* __restrict__ b, float* __restrict__ out) {
    int i = blockIdx.x * 256 + threadIdx.x;
    if (i >= BATCH * NCLS) return;
    int bb = i / NCLS, n = i % NCLS;
    const float* cr = cls + (size_t)bb * EMB;
    float acc = b[n];
    for (int k = 0; k < EMB; ++k) acc += cr[k] * w[(size_t)k * NCLS + n];
    out[i] = acc;
}

static inline unsigned nblk(long long n) {
    long long b = (n + 255) / 256;
    if (b > 8192) b = 8192;
    return (unsigned)b;
}

extern "C" void kernel_launch(void* const* d_in, const int* in_sizes, int n_in,
                              void* d_out, int out_size, void* d_ws, size_t ws_size,
                              hipStream_t stream)
{
    (void)in_sizes; (void)n_in; (void)out_size; (void)ws_size;
    const float* x       = (const float*)d_in[0];
    const float* w_patch = (const float*)d_in[1];
    const float* b_patch = (const float*)d_in[2];
    const float* cls_tok = (const float*)d_in[3];
    const float* pos_emb = (const float*)d_in[4];
    const float* ln1_w   = (const float*)d_in[5];
    const float* ln1_b   = (const float*)d_in[6];
    const float* wqkv    = (const float*)d_in[7];
    const float* bqkv    = (const float*)d_in[8];
    const float* wproj   = (const float*)d_in[9];
    const float* bproj   = (const float*)d_in[10];
    const float* ln2_w   = (const float*)d_in[11];
    const float* ln2_b   = (const float*)d_in[12];
    const float* wfc1    = (const float*)d_in[13];
    const float* bfc1    = (const float*)d_in[14];
    const float* wfc2    = (const float*)d_in[15];
    const float* bfc2    = (const float*)d_in[16];
    const float* lnf_w   = (const float*)d_in[17];
    const float* lnf_b   = (const float*)d_in[18];
    const float* w_head  = (const float*)d_in[19];
    const float* b_head  = (const float*)d_in[20];
    float* out = (float*)d_out;

    char* ws = (char*)d_ws;
    size_t off = 0;
    auto alloc = [&](size_t bytes) -> char* {
        char* p = ws + off;
        off = (off + bytes + 255) & ~(size_t)255;
        return p;
    };
    float*    hbuf  = (float*)   alloc(sizeof(float) * MTOT * EMB);
    float*    nbuf  = (float*)   alloc(sizeof(float) * MTOT * EMB);
    float*    qkvb  = (float*)   alloc(sizeof(float) * MTOT * 3 * EMB);
    float*    mbuf  = (float*)   alloc(sizeof(float) * MTOT * HID);
    float*    obuf  = (float*)   alloc(sizeof(float) * MTOT * EMB);
    int8_t*   qa8   = (int8_t*)  alloc((size_t)MTOT * HID);
    int8_t*   qw8t  = (int8_t*)  alloc((size_t)HID * EMB);
    unsigned* amaxA = (unsigned*)alloc(256);
    float*    sclA  = (float*)((char*)amaxA + 64);
    unsigned* amaxW = (unsigned*)((char*)amaxA + 128);
    float*    sclW  = (float*)((char*)amaxA + 192);
    _Float16* q16   = (_Float16*)alloc(sizeof(_Float16) * BH * SQ * HDIM);
    _Float16* k16   = (_Float16*)alloc(sizeof(_Float16) * BH * SK * HDIM);
    _Float16* vt16  = (_Float16*)alloc(sizeof(_Float16) * BH * HDIM * SK);
    float*    sc32  = (float*)   alloc(sizeof(float) * BH * SQ * SK);
    _Float16* pr16  = (_Float16*)alloc(sizeof(_Float16) * BH * SQ * SK);
    _Float16* pat16 = (_Float16*)alloc(sizeof(_Float16) * MPATCH * EMB);
    _Float16* wp16  = (_Float16*)alloc(sizeof(_Float16) * EMB * EMB);
    float*    cls32 = (float*)   alloc(sizeof(float) * BATCH * EMB);

    auto qlinear = [&](const float* act, long long m, long long k, long long n,
                       const float* w, const float* bias, float* outp,
                       const float* resid, int op) {
        long long na = m * k, nw = k * n;
        k_zero1 <<<1, 1, 0, stream>>>(amaxA);
        k_zero1 <<<1, 1, 0, stream>>>(amaxW);
        k_absmax<<<nblk(na), 256, 0, stream>>>(act, na, amaxA);
        k_absmax<<<nblk(nw), 256, 0, stream>>>(w,   nw, amaxW);
        k_quant <<<nblk(na), 256, 0, stream>>>(act, na, amaxA, sclA, qa8);
        k_quantT<<<nblk(nw), 256, 0, stream>>>(w, (int)k, (int)n, amaxW, sclW, qw8t);
        dim3 g((unsigned)(n / 128), (unsigned)((m + 127) / 128));
        k_gemm_i8<<<g, 256, 0, stream>>>(qa8, qw8t, sclA, sclW, bias, resid, outp,
                                         (int)m, (int)n, (int)k, op);
    };

    k_patchify<<<nblk((long long)MPATCH * EMB), 256, 0, stream>>>(x, pat16);
    k_f32tof16<<<nblk((long long)EMB * EMB), 256, 0, stream>>>(w_patch, wp16, (long long)EMB * EMB);
    {
        dim3 g(EMB / 16, MPATCH / 16, 1);
        k_gemm_f16<<<g, 32, 0, stream>>>(pat16, wp16, b_patch, nbuf,
                                         MPATCH, EMB, EMB, EMB, EMB, EMB,
                                         0, 0, 0, 0, 1);
    }
    k_assemble<<<nblk((long long)MTOT * EMB), 256, 0, stream>>>(nbuf, cls_tok, pos_emb, hbuf);

    for (int l = 0; l < 12; ++l) {
        const float* l1w = ln1_w + (size_t)l * EMB;
        const float* l1b = ln1_b + (size_t)l * EMB;
        const float* wq  = wqkv  + (size_t)l * EMB * 3 * EMB;
        const float* bq  = bqkv  + (size_t)l * 3 * EMB;
        const float* wp  = wproj + (size_t)l * EMB * EMB;
        const float* bp  = bproj + (size_t)l * EMB;
        const float* l2w = ln2_w + (size_t)l * EMB;
        const float* l2b = ln2_b + (size_t)l * EMB;
        const float* w1  = wfc1  + (size_t)l * EMB * HID;
        const float* b1  = bfc1  + (size_t)l * HID;
        const float* w2  = wfc2  + (size_t)l * HID * EMB;
        const float* b2  = bfc2  + (size_t)l * EMB;

        k_layernorm<<<MTOT, 256, 0, stream>>>(hbuf, l1w, l1b, nbuf, EMB, EMB);
        qlinear(nbuf, MTOT, EMB, 3 * EMB, wq, bq, qkvb, nullptr, 0);

        (void)hipMemsetAsync(q16,  0, sizeof(_Float16) * BH * SQ * HDIM, stream);
        (void)hipMemsetAsync(k16,  0, sizeof(_Float16) * BH * SK * HDIM, stream);
        (void)hipMemsetAsync(vt16, 0, sizeof(_Float16) * BH * HDIM * SK, stream);
        k_qkv_split<<<nblk((long long)MTOT * EMB), 256, 0, stream>>>(qkvb, q16, k16, vt16);
        {
            dim3 g(SQ / 16, SQ / 16, BH);
            k_gemm_f16<<<g, 32, 0, stream>>>(q16, k16, nullptr, sc32,
                                             SEQ, SQ, HDIM, HDIM, HDIM, SK,
                                             (long long)SQ * HDIM, (long long)SK * HDIM,
                                             (long long)SQ * SK, 0, 1);
        }
        k_softmax<<<BH * SQ, 64, 0, stream>>>(sc32, pr16);
        {
            dim3 g(HDIM / 16, SQ / 16, BH);
            k_gemm_f16<<<g, 32, 0, stream>>>(pr16, vt16, nullptr, obuf,
                                             SEQ, HDIM, SK, SK, SK, EMB,
                                             (long long)SQ * SK, (long long)HDIM * SK,
                                             (long long)SEQ * EMB, HDIM, NHEAD);
        }

        qlinear(obuf, MTOT, EMB, EMB, wp, bp, hbuf, hbuf, 2);

        k_layernorm<<<MTOT, 256, 0, stream>>>(hbuf, l2w, l2b, nbuf, EMB, EMB);
        qlinear(nbuf, MTOT, EMB, HID, w1, b1, mbuf, nullptr, 1);
        qlinear(mbuf, MTOT, HID, EMB, w2, b2, hbuf, hbuf, 2);
    }

    k_layernorm<<<BATCH, 256, 0, stream>>>(hbuf, lnf_w, lnf_b, cls32,
                                           (long long)SEQ * EMB, EMB);
    k_head<<<(BATCH * NCLS + 255) / 256, 256, 0, stream>>>(cls32, w_head, b_head, out);
}
